// RoiHead_54485955117628
// MI455X (gfx1250) — compile-verified, matrix operand order checked by reading, not run on hardware
//
#include <hip/hip_runtime.h>
#include <hip/hip_bf16.h>

// ---------------------------------------------------------------------------
// Problem constants (from reference)
// ---------------------------------------------------------------------------
#define NB   8
#define CC   528
#define TT   16
#define HH   32
#define WW   32
#define RR   1024
#define OUTB 8
#define SRR  2
#define SCL  (1.0f / 16.0f)

#define KPAD 544            // 528 padded to multiple of 32 for WMMA K
#define HID1 128
#define HID2 32
#define NPTS 256            // (OUTB*SRR)^2 sample points per roi
#define MAXC 12             // max roi footprint per axis in feature cells
#define CTILES 17           // ceil(528/32)

typedef __attribute__((ext_vector_type(16))) _Float16 v16h;
typedef __attribute__((ext_vector_type(8)))  _Float16 v8h;
typedef __attribute__((ext_vector_type(8)))  float    v8f;

// ---------------------------------------------------------------------------
// Kernel 1: fused temporal mean + NCHW->NHWC transpose.
//   x (N,C,T,H,W) f32  ->  featT (N,H*W,C) f32
// Reads 277 MB coalesced (contiguous across hw within a row, 16 t-strided
// rows per thread), transposes a 32c x 32hw tile through LDS, writes 17 MB
// coalesced along C. This is the HBM-bound phase (~12.7 us at 23.3 TB/s).
// ---------------------------------------------------------------------------
__global__ __launch_bounds__(256) void k_tmean_t(const float* __restrict__ x,
                                                 float* __restrict__ featT) {
    __shared__ float tile[32][33];
    const int b   = blockIdx.x;                 // NB * CTILES * 32 blocks
    const int n   = b / (CTILES * 32);
    const int rem = b % (CTILES * 32);
    const int ct  = rem / 32;                   // channel tile 0..16
    const int hwt = rem % 32;                   // hw tile 0..31
    const int c0  = ct * 32, hw0 = hwt * 32;
    const int tx  = threadIdx.x & 31;
    const int ty  = threadIdx.x >> 5;           // 32x8 thread tile

    #pragma unroll
    for (int s = 0; s < 4; ++s) {
        const int cy = ty + 8 * s;
        const int c  = c0 + cy;
        float sum = 0.0f;
        if (c < CC) {
            const float* p = x + (((size_t)(n * CC + c) * TT) << 10) + hw0 + tx;
            #pragma unroll
            for (int t = 0; t < TT; ++t) sum += p[t << 10];
        }
        tile[cy][tx] = sum * (1.0f / (float)TT);
    }
    __syncthreads();
    #pragma unroll
    for (int s = 0; s < 4; ++s) {
        const int hwy = ty + 8 * s;
        const int c   = c0 + tx;
        if (c < CC)
            featT[((size_t)((n << 10) + hw0 + hwy)) * CC + c] = tile[tx][hwy];
    }
}

// ---------------------------------------------------------------------------
// Kernel 2: transpose + f16-convert weights.
//   W1 (528,128) -> W1t f16 (128,544) zero-padded K
//   W2 (128,32)  -> W2t f16 (32,128)
// ---------------------------------------------------------------------------
__global__ __launch_bounds__(256) void k_prep(const float* __restrict__ W1,
                                              const float* __restrict__ W2,
                                              _Float16* __restrict__ W1t,
                                              _Float16* __restrict__ W2t) {
    int i = blockIdx.x * 256 + threadIdx.x;
    if (i < HID1 * KPAD) {
        int n = i / KPAD, k = i - n * KPAD;
        W1t[i] = (k < CC) ? (_Float16)W1[k * HID1 + n] : (_Float16)0.0f;
    } else {
        int j = i - HID1 * KPAD;
        if (j < HID2 * HID1) {
            int n = j / HID1, k = j - n * HID1;
            W2t[j] = (_Float16)W2[k * HID2 + n];
        }
    }
}

// ---------------------------------------------------------------------------
// Kernel 3: ROI-align + spatial mean -> pooled f16 (R, KPAD).
// One block per roi. The 256 bilinear sample points collapse into a <=12x12
// per-cell weight map (LDS, ds_add_f32 atomics, includes mask + 1/256 mean).
// Channel gather is NHWC: per cell the block reads a contiguous 2KB channel
// vector (float2/lane, global_load_b64 coalesced) scaled by the uniform
// cell weight. Feature data stays L2-resident across rois.
// ---------------------------------------------------------------------------
__device__ __forceinline__ float clampf(float v, float lo, float hi) {
    return fminf(fmaxf(v, lo), hi);
}

__global__ __launch_bounds__(256) void k_roipool(const float* __restrict__ featT,
                                                 const float* __restrict__ bbox,
                                                 _Float16* __restrict__ pooled) {
    __shared__ float wcell[MAXC * MAXC];
    const int r   = blockIdx.x;
    const int tid = threadIdx.x;

    if (tid < MAXC * MAXC) wcell[tid] = 0.0f;

    const float* bb = bbox + r * 5;
    const int   bi  = (int)bb[0];
    const float sx1 = bb[1] * SCL - 0.5f;
    const float sy1 = bb[2] * SCL - 0.5f;
    const float sx2 = bb[3] * SCL - 0.5f;
    const float sy2 = bb[4] * SCL - 0.5f;
    const float bws = (sx2 - sx1) * (1.0f / (OUTB * SRR));   // binw / SR
    const float bhs = (sy2 - sy1) * (1.0f / (OUTB * SRR));   // binh / SR

    // footprint origin / extent (monotone in sample index since bw,bh > 0)
    const int oy = (int)floorf(clampf(sy1 + 0.5f  * bhs, 0.f, HH - 1.f));
    const int ox = (int)floorf(clampf(sx1 + 0.5f  * bws, 0.f, WW - 1.f));
    int ey = min((int)floorf(clampf(sy1 + 15.5f * bhs, 0.f, HH - 1.f)) + 1, HH - 1);
    int ex = min((int)floorf(clampf(sx1 + 15.5f * bws, 0.f, WW - 1.f)) + 1, WW - 1);
    int hy = min(ey - oy + 1, MAXC);
    int wx = min(ex - ox + 1, MAXC);

    __syncthreads();

    // ---- scatter bilinear weights for this thread's sample point ----
    {
        const int iy = tid >> 4, ix = tid & 15;
        const float yv = sy1 + ((float)iy + 0.5f) * bhs;
        const float xv = sx1 + ((float)ix + 0.5f) * bws;
        const bool vy = (yv >= -1.0f) && (yv <= (float)HH);
        const bool vx = (xv >= -1.0f) && (xv <= (float)WW);
        const float yc = clampf(yv, 0.f, HH - 1.f);
        const float xc = clampf(xv, 0.f, WW - 1.f);
        int y0 = (int)floorf(yc); float ly = yc - (float)y0;
        int x0 = (int)floorf(xc); float lx = xc - (float)x0;
        int y1 = min(y0 + 1, HH - 1);
        int x1 = min(x0 + 1, WW - 1);
        const float m = (vy && vx) ? (1.0f / (float)NPTS) : 0.0f;
        // clamp relative cell coords into the LDS box (memory safety)
        int cy0 = min(max(y0 - oy, 0), hy - 1);
        int cy1 = min(max(y1 - oy, 0), hy - 1);
        int cx0 = min(max(x0 - ox, 0), wx - 1);
        int cx1 = min(max(x1 - ox, 0), wx - 1);
        atomicAdd(&wcell[cy0 * wx + cx0], (1.f - ly) * (1.f - lx) * m);
        atomicAdd(&wcell[cy0 * wx + cx1], (1.f - ly) * lx * m);
        atomicAdd(&wcell[cy1 * wx + cx0], ly * (1.f - lx) * m);
        atomicAdd(&wcell[cy1 * wx + cx1], ly * lx * m);
    }
    __syncthreads();

    // ---- coalesced NHWC channel gather over the cell map ----
    const float* fb = featT + ((size_t)(bi << 10)) * CC;
    const int c0 = 2 * tid;                 // channels 0..511
    const int c1 = 512 + 2 * tid;           // channels 512..527 (tid < 8)
    float a0x = 0.f, a0y = 0.f, a1x = 0.f, a1y = 0.f;
    for (int cy = 0; cy < hy; ++cy) {
        const size_t rowb = (size_t)((oy + cy) * WW + ox) * CC;
        for (int cx = 0; cx < wx; ++cx) {
            const float wv = wcell[cy * wx + cx];
            const float* cell = fb + rowb + (size_t)cx * CC;
            const float2 v0 = *(const float2*)(cell + c0);
            a0x = fmaf(wv, v0.x, a0x);
            a0y = fmaf(wv, v0.y, a0y);
            if (tid < 8) {
                const float2 v1 = *(const float2*)(cell + c1);
                a1x = fmaf(wv, v1.x, a1x);
                a1y = fmaf(wv, v1.y, a1y);
            }
        }
    }
    _Float16* pr = pooled + (size_t)r * KPAD;
    pr[c0]     = (_Float16)a0x;
    pr[c0 + 1] = (_Float16)a0y;
    if (tid < 8) {
        pr[c1]     = (_Float16)a1x;
        pr[c1 + 1] = (_Float16)a1y;
    }
    if (tid < KPAD - CC) pr[CC + tid] = (_Float16)0.0f;   // K padding
}

// ---------------------------------------------------------------------------
// WMMA fragment loader: 16-bit 16xK fragment, documented CDNA5 layout.
// lane&15 selects the row (A) / column (B-transposed); lanes 16-31 take the
// +8 half-dword K group. Two 16-byte loads fused into a v16h.
// ---------------------------------------------------------------------------
__device__ __forceinline__ v16h load_frag16(const _Float16* __restrict__ p, int lane) {
    p += (lane & 16) ? 8 : 0;
    v8h lo = *(const v8h*)p;
    v8h hi = *(const v8h*)(p + 16);
    return __builtin_shufflevector(lo, hi, 0, 1, 2, 3, 4, 5, 6, 7,
                                           8, 9, 10, 11, 12, 13, 14, 15);
}

// ---------------------------------------------------------------------------
// Kernel 4: GEMM1  h1 = relu(pooled @ W1 + b1)
//   M=1024, N=128, K=544 (f16 in, f32 acc, f16 out). 1 wave = one 16x16 tile.
// ---------------------------------------------------------------------------
__global__ __launch_bounds__(256) void k_gemm1(const _Float16* __restrict__ A,
                                               const _Float16* __restrict__ Bt,
                                               const float* __restrict__ bias,
                                               _Float16* __restrict__ Hout) {
    const int lane = threadIdx.x & 31;
    const int wid  = threadIdx.x >> 5;
    const int tile = blockIdx.x * 8 + wid;           // 512 tiles
    const int tm = tile >> 3, tn = tile & 7;
    const _Float16* arow = A  + (size_t)(tm * 16 + (lane & 15)) * KPAD;
    const _Float16* brow = Bt + (size_t)(tn * 16 + (lane & 15)) * KPAD;
    v8f c = {};
    #pragma unroll
    for (int kk = 0; kk < KPAD; kk += 32) {
        v16h a = load_frag16(arow + kk, lane);
        v16h b = load_frag16(brow + kk, lane);
        c = __builtin_amdgcn_wmma_f32_16x16x32_f16(false, a, false, b,
                                                   (short)0, c, false, false);
    }
    const float bn   = bias[tn * 16 + (lane & 15)];
    const int mbase  = tm * 16 + ((lane >> 4) << 3);
    const int ncol   = tn * 16 + (lane & 15);
    #pragma unroll
    for (int q = 0; q < 8; ++q) {
        float v = c[q] + bn;
        v = v > 0.0f ? v : 0.0f;                     // ReLU
        Hout[(size_t)(mbase + q) * HID1 + ncol] = (_Float16)v;
    }
}

// ---------------------------------------------------------------------------
// Kernel 5: GEMM2  h2 = h1 @ W2 + b2   (M=1024, N=32, K=128, f32 out)
// ---------------------------------------------------------------------------
__global__ __launch_bounds__(256) void k_gemm2(const _Float16* __restrict__ A,
                                               const _Float16* __restrict__ Bt,
                                               const float* __restrict__ bias,
                                               float* __restrict__ Hout) {
    const int lane = threadIdx.x & 31;
    const int wid  = threadIdx.x >> 5;
    const int tile = blockIdx.x * 8 + wid;           // 128 tiles
    const int tm = tile >> 1, tn = tile & 1;
    const _Float16* arow = A  + (size_t)(tm * 16 + (lane & 15)) * HID1;
    const _Float16* brow = Bt + (size_t)(tn * 16 + (lane & 15)) * HID1;
    v8f c = {};
    #pragma unroll
    for (int kk = 0; kk < HID1; kk += 32) {
        v16h a = load_frag16(arow + kk, lane);
        v16h b = load_frag16(brow + kk, lane);
        c = __builtin_amdgcn_wmma_f32_16x16x32_f16(false, a, false, b,
                                                   (short)0, c, false, false);
    }
    const float bn  = bias[tn * 16 + (lane & 15)];
    const int mbase = tm * 16 + ((lane >> 4) << 3);
    const int ncol  = tn * 16 + (lane & 15);
    #pragma unroll
    for (int q = 0; q < 8; ++q)
        Hout[(size_t)(mbase + q) * HID2 + ncol] = c[q] + bn;
}

// ---------------------------------------------------------------------------
// Kernel 6: head  out = sigmoid(h2 @ W3 + b3)
// ---------------------------------------------------------------------------
__global__ __launch_bounds__(256) void k_head(const float* __restrict__ h2,
                                              const float* __restrict__ W3,
                                              const float* __restrict__ b3,
                                              float* __restrict__ out) {
    int r = blockIdx.x * 256 + threadIdx.x;
    if (r >= RR) return;
    const float* hr = h2 + (size_t)r * HID2;
    float acc = b3[0];
    #pragma unroll
    for (int j = 0; j < HID2; ++j) acc = fmaf(hr[j], W3[j], acc);
    out[r] = 1.0f / (1.0f + __expf(-acc));
}

// ---------------------------------------------------------------------------
// Launcher
// ---------------------------------------------------------------------------
extern "C" void kernel_launch(void* const* d_in, const int* in_sizes, int n_in,
                              void* d_out, int out_size, void* d_ws, size_t ws_size,
                              hipStream_t stream) {
    const float* x    = (const float*)d_in[0];
    const float* bbox = (const float*)d_in[1];
    const float* W1   = (const float*)d_in[2];
    const float* b1   = (const float*)d_in[3];
    const float* W2   = (const float*)d_in[4];
    const float* b2   = (const float*)d_in[5];
    const float* W3   = (const float*)d_in[6];
    const float* b3   = (const float*)d_in[7];
    float* out = (float*)d_out;

    // workspace layout (all sizes multiples of 256B)
    char* ws = (char*)d_ws;
    size_t off = 0;
    float*    featT   = (float*)(ws + off);     off += (size_t)NB * HH * WW * CC * 4;   // 17.3 MB (NHWC)
    _Float16* pooled  = (_Float16*)(ws + off);  off += (size_t)RR * KPAD * 2;           // 1.1 MB
    _Float16* W1t     = (_Float16*)(ws + off);  off += (size_t)HID1 * KPAD * 2;         // 139 KB
    _Float16* W2t     = (_Float16*)(ws + off);  off += (size_t)HID2 * HID1 * 2;         // 8 KB
    _Float16* h1      = (_Float16*)(ws + off);  off += (size_t)RR * HID1 * 2;           // 256 KB
    float*    h2      = (float*)(ws + off);     off += (size_t)RR * HID2 * 4;           // 128 KB

    // 1) fused temporal mean + transpose to NHWC: 8 * 17 * 32 = 4352 blocks
    k_tmean_t<<<NB * CTILES * 32, 256, 0, stream>>>(x, featT);

    // 2) weight transpose + f16 convert
    {
        int total = HID1 * KPAD + HID2 * HID1;      // 73,728
        k_prep<<<(total + 255) / 256, 256, 0, stream>>>(W1, W2, W1t, W2t);
    }

    // 3) roi-align + pool -> pooled f16 (one block per roi)
    k_roipool<<<RR, 256, 0, stream>>>(featT, bbox, pooled);

    // 4) GEMM1 (WMMA): 512 tiles, 8 waves/block -> 64 blocks
    k_gemm1<<<64, 256, 0, stream>>>(pooled, W1t, b1, h1);

    // 5) GEMM2 (WMMA): 128 tiles -> 16 blocks
    k_gemm2<<<16, 256, 0, stream>>>(h1, W2t, b2, h2);

    // 6) head
    k_head<<<RR / 256, 256, 0, stream>>>(h2, W3, b3, out);
}
